// DynamicConvDecoderLayer_60790967107734
// MI455X (gfx1250) — compile-verified, matrix operand order checked by reading, not run on hardware
//
#include <hip/hip_runtime.h>

// ---------------------------------------------------------------------------
// DynamicConv decoder layer for MI455X (gfx1250, wave32, WMMA, async-LDS).
// GEMMs: bf16 WMMA (v_wmma_f32_16x16x32_bf16), f32 accumulation, double-
// buffered LDS staged with global_load_async_to_lds_b128 / s_wait_asynccnt.
// ---------------------------------------------------------------------------

typedef __attribute__((ext_vector_type(16))) __bf16        v16bf;
typedef __attribute__((ext_vector_type(8)))  float         v8f;
typedef __attribute__((ext_vector_type(4)))  unsigned int  uint4v;

#define TDIM   2048
#define BDIM   16
#define EDIM   1024
#define CDIMC  1024
#define HEADS  16
#define KW     31
#define MROWS  (TDIM * BDIM)   /* 32768 */
#define NWPAD  512             /* HEADS*KW = 496 padded to 512 */
#define RCH    (CDIMC / HEADS) /* 64 channels per head */
#define GK     1024            /* K depth of every GEMM here */

__device__ __forceinline__ unsigned short f2bf(float f) {
  unsigned int u = __float_as_uint(f);
  u += 0x7FFFu + ((u >> 16) & 1u);          // round-to-nearest-even
  return (unsigned short)(u >> 16);
}
__device__ __forceinline__ unsigned int pack2bf(float a, float b) {
  return (unsigned int)f2bf(a) | ((unsigned int)f2bf(b) << 16);
}
__device__ __forceinline__ float bf2f(unsigned short h) {
  return __uint_as_float((unsigned int)h << 16);
}

// issue one 16-byte async global->LDS copy (per-lane, tracked by ASYNCcnt)
__device__ __forceinline__ void async_cp16(unsigned lds_addr, const unsigned short* g) {
  asm volatile("global_load_async_to_lds_b128 %0, %1, off"
               :: "v"(lds_addr), "v"((unsigned long long)g)
               : "memory");
}
// low 32 bits of a generic shared pointer == LDS byte address (aperture layout)
__device__ __forceinline__ unsigned lds_lo32(const void* p) {
  return (unsigned)(unsigned long long)p;
}

// ---------------------------------------------------------------------------
// elementwise f32 -> bf16 (pairs)
// ---------------------------------------------------------------------------
__global__ __launch_bounds__(256) void dcl_f32_to_bf16(
    const float* __restrict__ in, unsigned int* __restrict__ out, long npairs) {
  long i = (long)blockIdx.x * 256 + threadIdx.x;
  if (i >= npairs) return;
  float2 v = ((const float2*)in)[i];
  out[i] = pack2bf(v.x, v.y);
}

// weight conversion f32 [rows_in, cols] -> bf16 [rows_out, cols], zero row pad
__global__ void dcl_convert_pad_bf16(const float* __restrict__ in,
                                     unsigned short* __restrict__ out,
                                     int rows_in, long total, int cols) {
  long i = (long)blockIdx.x * blockDim.x + threadIdx.x;
  if (i >= total) return;
  long r = i / cols;
  out[i] = f2bf(r < rows_in ? in[i] : 0.0f);
}

__global__ void dcl_pad_bias(const float* __restrict__ in, float* __restrict__ out,
                             int n_in, int n_out) {
  int i = blockIdx.x * blockDim.x + threadIdx.x;
  if (i < n_out) out[i] = (i < n_in) ? in[i] : 0.0f;
}

// ---------------------------------------------------------------------------
// WMMA GEMM: C[M,LDC] = A[M,GK](bf16) * Bw[LDC,GK](bf16)^T + bias
// Block tile 128x128, K-step 32, 256 threads = 8 waves (4 along M x 2 along N),
// wave tile 32x64 -> 2x4 subtiles -> 8 v_wmma per K-step.
// Double-buffered LDS filled by global_load_async_to_lds_b128.
// ---------------------------------------------------------------------------
#define GBM 128
#define GBN 128
#define GBK 32
#define LDST 40   // LDS row stride in ushorts: 80B => bank-conflict-free, 16B aligned

template <int LDC, bool BF16OUT>
__global__ __launch_bounds__(256) void dcl_wmma_gemm(
    const unsigned short* __restrict__ A, const unsigned short* __restrict__ Bw,
    const float* __restrict__ bias, float* __restrict__ Cf,
    unsigned short* __restrict__ Cb) {
  __shared__ unsigned short sA[2][GBM * LDST];
  __shared__ unsigned short sB[2][GBN * LDST];

  const int tid  = threadIdx.x;
  const int lane = tid & 31;
  const int wave = tid >> 5;
  const int wm = wave >> 1;             // 0..3 along M (32-row wave tile)
  const int wn = wave & 1;              // 0..1 along N (64-col wave tile)

  const int m0 = blockIdx.y * GBM;
  const int n0 = blockIdx.x * GBN;

  // each tile = 128 rows x 4 16B-chunks; thread owns chunks {tid, tid+256}
  const int r0 = tid >> 2,            c0 = (tid & 3) << 3;
  const int r1 = (tid + 256) >> 2,    c1 = ((tid + 256) & 3) << 3;

  // WMMA 16-bit operand striping (ISA 7.12.2)
  const int kl0  = (lane < 16) ? 0 : 8;
  const int fidx = lane & 15;

  v8f acc[2][4] = {};

  // preload bias (per-lane, 4 N-subtiles)
  float bv[4];
#pragma unroll
  for (int j = 0; j < 4; ++j) bv[j] = bias[n0 + wn * 64 + j * 16 + fidx];

  auto issue = [&](int buf, int kk) {
    async_cp16(lds_lo32(&sA[buf][r0 * LDST + c0]), A  + (size_t)(m0 + r0) * GK + kk + c0);
    async_cp16(lds_lo32(&sA[buf][r1 * LDST + c1]), A  + (size_t)(m0 + r1) * GK + kk + c1);
    async_cp16(lds_lo32(&sB[buf][r0 * LDST + c0]), Bw + (size_t)(n0 + r0) * GK + kk + c0);
    async_cp16(lds_lo32(&sB[buf][r1 * LDST + c1]), Bw + (size_t)(n0 + r1) * GK + kk + c1);
  };

  issue(0, 0);
  int buf = 0;
  for (int kk = 0; kk < GK; kk += GBK, buf ^= 1) {
    if (kk + GBK < GK) {
      issue(buf ^ 1, kk + GBK);
      // in-order async completion: <=4 outstanding => tile `buf` has landed
      asm volatile("s_wait_asynccnt 0x4" ::: "memory");
    } else {
      asm volatile("s_wait_asynccnt 0x0" ::: "memory");
    }
    __syncthreads();

    union Frag { uint4v u[2]; v16bf v; };
    Frag a[2], b[4];
#pragma unroll
    for (int i = 0; i < 2; ++i) {
      const unsigned short* pa = &sA[buf][(wm * 32 + i * 16 + fidx) * LDST + kl0];
      a[i].u[0] = *(const uint4v*)(pa);
      a[i].u[1] = *(const uint4v*)(pa + 16);
    }
#pragma unroll
    for (int j = 0; j < 4; ++j) {
      const unsigned short* pb = &sB[buf][(wn * 64 + j * 16 + fidx) * LDST + kl0];
      b[j].u[0] = *(const uint4v*)(pb);
      b[j].u[1] = *(const uint4v*)(pb + 16);
    }
#pragma unroll
    for (int i = 0; i < 2; ++i)
#pragma unroll
      for (int j = 0; j < 4; ++j)
        acc[i][j] = __builtin_amdgcn_wmma_f32_16x16x32_bf16(
            false, a[i].v, false, b[j].v, (short)0, acc[i][j], false, false);
    __syncthreads();
  }

  // epilogue: D layout: VGPR r, lane l -> M = r + (l<16?0:8), N = l&15
  const int radd = (lane < 16) ? 0 : 8;
#pragma unroll
  for (int i = 0; i < 2; ++i) {
#pragma unroll
    for (int j = 0; j < 4; ++j) {
      const int col  = n0 + wn * 64 + j * 16 + fidx;
      const int row0 = m0 + wm * 32 + i * 16 + radd;
      if (BF16OUT) {
        unsigned short* p = Cb + (size_t)row0 * LDC + col;
#pragma unroll
        for (int r = 0; r < 8; ++r) p[r * LDC] = f2bf(acc[i][j][r] + bv[j]);
      } else {
        float* p = Cf + (size_t)row0 * LDC + col;
#pragma unroll
        for (int r = 0; r < 8; ++r) p[r * LDC] = acc[i][j][r] + bv[j];
      }
    }
  }
}

// ---------------------------------------------------------------------------
// Softmax over K=31 per (row, head); one wave32 per group; u is [M, 512] f32
// ---------------------------------------------------------------------------
__global__ __launch_bounds__(256) void dcl_softmax31(float* __restrict__ u) {
  const int g    = blockIdx.x * 8 + (threadIdx.x >> 5);
  const int lane = threadIdx.x & 31;
  const int m = g >> 4;
  const int h = g & 15;
  float* p = u + (size_t)m * NWPAD + h * KW;
  float v = (lane < KW) ? p[lane] : -3.402823466e38f;
  float mx = v;
#pragma unroll
  for (int off = 16; off; off >>= 1) mx = fmaxf(mx, __shfl_xor(mx, off, 32));
  float e = (lane < KW) ? __expf(v - mx) : 0.0f;
  float s = e;
#pragma unroll
  for (int off = 16; off; off >>= 1) s += __shfl_xor(s, off, 32);
  if (lane < KW) p[lane] = e / s;
}

// ---------------------------------------------------------------------------
// Causal dynamic conv on bf16 h: out_bf16[t,b,h*64+r] =
//   sum_k w[t,b,h,k] * h[t+k-30, b, h*64+r]; LDS-staged tile + halo.
// ---------------------------------------------------------------------------
__global__ __launch_bounds__(256) void dcl_dynconv(
    const unsigned short* __restrict__ hin, const float* __restrict__ w,
    unsigned short* __restrict__ out) {
  __shared__ float hs[64 + KW - 1][RCH];   // 94 x 64 f32
  __shared__ float wg[64][KW];

  const int t0  = blockIdx.x * 64;
  const int b   = blockIdx.y;
  const int hd  = blockIdx.z;
  const int tid = threadIdx.x;

  for (int idx = tid; idx < (64 + KW - 1) * RCH; idx += 256) {
    const int j = idx >> 6, r = idx & 63;
    const int t = t0 - (KW - 1) + j;
    hs[j][r] = (t >= 0)
        ? bf2f(hin[((size_t)t * BDIM + b) * CDIMC + hd * RCH + r]) : 0.0f;
  }
  for (int idx = tid; idx < 64 * KW; idx += 256) {
    const int tl = idx / KW, k = idx % KW;
    wg[tl][k] = w[((size_t)(t0 + tl) * BDIM + b) * NWPAD + hd * KW + k];
  }
  __syncthreads();

  const int r  = tid & 63;
  const int tg = tid >> 6;      // 0..3
#pragma unroll
  for (int i = 0; i < 16; ++i) {
    const int tl = tg * 16 + i;
    float acc = 0.0f;
#pragma unroll
    for (int k = 0; k < KW; ++k) acc += wg[tl][k] * hs[tl + k][r];
    out[((size_t)(t0 + tl) * BDIM + b) * CDIMC + hd * RCH + r] = f2bf(acc);
  }
}

// ---------------------------------------------------------------------------
// Residual + LayerNorm (in place on y = d_out): one 256-thread WG per row
// ---------------------------------------------------------------------------
__global__ __launch_bounds__(256) void dcl_residual_ln(
    const float* __restrict__ x, const float* __restrict__ gamma,
    const float* __restrict__ beta, float* __restrict__ y) {
  const int m = blockIdx.x;
  const size_t base = (size_t)m * EDIM;
  const int tid = threadIdx.x;

  float v[4];
  float s = 0.0f, sq = 0.0f;
#pragma unroll
  for (int i = 0; i < 4; ++i) {
    const int c = tid + 256 * i;
    const float t = x[base + c] + y[base + c];
    v[i] = t; s += t; sq += t * t;
  }
#pragma unroll
  for (int off = 16; off; off >>= 1) {
    s  += __shfl_xor(s,  off, 32);
    sq += __shfl_xor(sq, off, 32);
  }
  __shared__ float rs[8], rq[8];
  const int wv = tid >> 5, lane = tid & 31;
  if (lane == 0) { rs[wv] = s; rq[wv] = sq; }
  __syncthreads();
  float ts = 0.0f, tq = 0.0f;
#pragma unroll
  for (int j = 0; j < 8; ++j) { ts += rs[j]; tq += rq[j]; }
  const float mean = ts * (1.0f / EDIM);
  const float var  = tq * (1.0f / EDIM) - mean * mean;
  const float inv  = rsqrtf(var + 1e-5f);
#pragma unroll
  for (int i = 0; i < 4; ++i) {
    const int c = tid + 256 * i;
    y[base + c] = (v[i] - mean) * inv * gamma[c] + beta[c];
  }
}

// ---------------------------------------------------------------------------
extern "C" void kernel_launch(void* const* d_in, const int* in_sizes, int n_in,
                              void* d_out, int out_size, void* d_ws, size_t ws_size,
                              hipStream_t stream) {
  (void)in_sizes; (void)n_in; (void)out_size; (void)ws_size;

  const float* x     = (const float*)d_in[0];
  const float* w1    = (const float*)d_in[1];
  const float* b1    = (const float*)d_in[2];
  const float* ww    = (const float*)d_in[3];
  const float* bw    = (const float*)d_in[4];
  const float* w2    = (const float*)d_in[5];
  const float* b2    = (const float*)d_in[6];
  const float* gamma = (const float*)d_in[7];
  const float* beta  = (const float*)d_in[8];
  float* out = (float*)d_out;

  // workspace layout (~261 MB)
  char* ws = (char*)d_ws;
  const size_t UB = (size_t)MROWS * NWPAD * sizeof(float);        // 64 MB
  const size_t XB = (size_t)MROWS * CDIMC * sizeof(unsigned short); // 64 MB
  float*          ubuf = (float*)(ws);                 // conv weights [M,512] f32
  unsigned short* xb   = (unsigned short*)(ws + UB);   // bf16(x)
  unsigned short* hb   = (unsigned short*)(ws + UB + XB);       // bf16 linear1 out
  unsigned short* cb   = (unsigned short*)(ws + UB + 2 * XB);   // bf16 conv out
  unsigned short* w1b  = (unsigned short*)(ws + UB + 3 * XB);
  unsigned short* w2b  = w1b + (size_t)CDIMC * EDIM;
  unsigned short* wwb  = w2b + (size_t)EDIM * CDIMC;
  float*          bwp  = (float*)(wwb + (size_t)NWPAD * CDIMC);

  // 1) one-shot conversions to bf16
  {
    const long npx = (long)MROWS * EDIM / 2;
    dcl_f32_to_bf16<<<(npx + 255) / 256, 256, 0, stream>>>(x, (unsigned int*)xb, npx);
    const long tot = (long)CDIMC * EDIM;
    dcl_convert_pad_bf16<<<(tot + 255) / 256, 256, 0, stream>>>(w1, w1b, CDIMC, tot, EDIM);
    dcl_convert_pad_bf16<<<(tot + 255) / 256, 256, 0, stream>>>(w2, w2b, EDIM, tot, CDIMC);
    const long tot2 = (long)NWPAD * CDIMC;
    dcl_convert_pad_bf16<<<(tot2 + 255) / 256, 256, 0, stream>>>(ww, wwb, HEADS * KW, tot2, CDIMC);
    dcl_pad_bias<<<2, 256, 0, stream>>>(bw, bwp, HEADS * KW, NWPAD);
  }

  // 2) linear1: hb = bf16(x @ w1^T + b1)
  dcl_wmma_gemm<CDIMC, true><<<dim3(CDIMC / GBN, MROWS / GBM), 256, 0, stream>>>(
      xb, w1b, b1, nullptr, hb);

  // 3) weight-linear: u = h @ ww^T + bw  (padded N=512, f32 out)
  dcl_wmma_gemm<NWPAD, false><<<dim3(NWPAD / GBN, MROWS / GBM), 256, 0, stream>>>(
      hb, wwb, bwp, ubuf, nullptr);

  // 4) softmax over K=31 groups (in place)
  dcl_softmax31<<<(MROWS * HEADS) / 8, 256, 0, stream>>>(ubuf);

  // 5) dynamic conv -> bf16
  dcl_dynconv<<<dim3(TDIM / 64, BDIM, HEADS), 256, 0, stream>>>(hb, ubuf, cb);

  // 6) linear2: y = conv @ w2^T + b2  -> d_out (f32)
  dcl_wmma_gemm<EDIM, false><<<dim3(EDIM / GBN, MROWS / GBM), 256, 0, stream>>>(
      cb, w2b, b2, out, nullptr);

  // 7) residual + LayerNorm (in place on d_out)
  dcl_residual_ln<<<MROWS, 256, 0, stream>>>(x, gamma, beta, out);
}
